// Diffusion_4294967296642
// MI455X (gfx1250) — matrix operand order repaired and checked by reading, not compiled, not run on hardware
//
#include <hip/hip_runtime.h>

// ---- CDNA5 WMMA vector types (wave32) ----
typedef __attribute__((ext_vector_type(16))) __bf16 v16bf;
typedef __attribute__((ext_vector_type(8)))  __bf16 v8bf;
typedef __attribute__((ext_vector_type(8)))  float  v8f;

// Problem constants (match setup_inputs: N=8, D=256, H=W=128, n_heads=8, ksize=5, dilation=1)
#define NB     8
#define DD     256
#define HH     128
#define WW     128
#define NHEADS 8
#define HD     32            // head dim = 256/8
#define HWSZ   (HH*WW)

// Tiling
#define TS   16              // 16x16 output tile per workgroup
#define SR   20              // padded source region (TS+4)
#define KR   24              // key region (TS+8)
#define NSRC (SR*SR)         // 400
#define NKEY (KR*KR)         // 576
#define ATT_STRIDE 26        // 25 window weights, padded

// LDS layout (bytes, all 64B aligned)
#define OFF_K 0
#define OFF_Q (OFF_K + NKEY*HD*2)          // 36864
#define OFF_V (OFF_Q + NSRC*HD*2)          // 62464
#define OFF_A (OFF_V + HD*NSRC*2)          // 88064
#define OFF_S (OFF_A + NSRC*ATT_STRIDE*2)  // 108864 : 8 waves x 16x64 f32 scores, aliased w/ out staging
#define SMEM_BYTES (OFF_S + 8*16*64*4)     // 141632 total -> 2 workgroups / WGP (320KB LDS)

#define SHUF16(a,b) __builtin_shufflevector(a, b, 0,1,2,3,4,5,6,7,8,9,10,11,12,13,14,15)

__global__ __launch_bounds__(256)
void natt_diffuse_kernel(const float* __restrict__ Vp, const float* __restrict__ Kp,
                         const float* __restrict__ Qp, const float* __restrict__ Mp,
                         float* __restrict__ Op) {
  extern __shared__ char smem[];
  __bf16* Ks = (__bf16*)(smem + OFF_K);   // [key_row(24x24)][c]   pixel-major
  __bf16* Qs = (__bf16*)(smem + OFF_Q);   // [src_row(20x20)][c]   pixel-major
  __bf16* Vt = (__bf16*)(smem + OFF_V);   // [c][src_row(20x20)]   channel-major
  __bf16* At = (__bf16*)(smem + OFF_A);   // [src_row][25] softmax*mask weights

  const int tid  = threadIdx.x;
  const int lane = tid & 31, wv = tid >> 5;
  const int hi   = lane >> 4, ln = lane & 15;

  const int bid    = blockIdx.x;
  const int tileId = bid & 63;            // 8x8 tiles of 16x16
  const int nhid   = bid >> 6;            // (n, head)
  const int nb = nhid >> 3, head = nhid & 7;
  const int ty0 = (tileId >> 3) * TS, tx0 = (tileId & 7) * TS;

  const float* Kg = Kp + (size_t)(nb*DD + head*HD) * HWSZ;
  const float* Qg = Qp + (size_t)(nb*DD + head*HD) * HWSZ;
  const float* Vg = Vp + (size_t)(nb*DD + head*HD) * HWSZ;

  // ---------------- Phase 0: stage Q/K/V tiles to LDS as bf16 (zero-pad OOB) ----------------
  for (int e = tid; e < NKEY*HD; e += 256) {
    int c = e / NKEY, r = e - c*NKEY;
    int gy = ty0 - 4 + r / KR, gx = tx0 - 4 + (r % KR);
    float v = (gy>=0 && gy<HH && gx>=0 && gx<WW) ? Kg[(size_t)c*HWSZ + gy*WW + gx] : 0.f;
    Ks[r*HD + c] = (__bf16)v;
  }
  for (int e = tid; e < NSRC*HD; e += 256) {
    int c = e / NSRC, r = e - c*NSRC;
    int gy = ty0 - 2 + r / SR, gx = tx0 - 2 + (r % SR);
    bool ib = (gy>=0 && gy<HH && gx>=0 && gx<WW);
    size_t g = (size_t)c*HWSZ + gy*WW + gx;
    Qs[r*HD + c]   = (__bf16)(ib ? Qg[g] : 0.f);
    Vt[c*NSRC + r] = (__bf16)(ib ? Vg[g] : 0.f);
  }
  __syncthreads();

  // ---------------- Phase 1: scores (WMMA) + softmax over 25 window positions ----------------
  float* Ssm = (float*)(smem + OFF_S) + wv * (16*64);   // per-wave 16x64 score scratch

  for (int blk = wv; blk < 25; blk += 8) {              // 25 source blocks of 4x4
    const int bly = (blk/5)*4, blx = (blk%5)*4;         // block origin in padded source coords

    // A operand: Q rows. Lane<16: K={0..7,16..23}; lane>=16: K={8..15,24..31}
    const int srow = (bly + (ln>>2))*SR + blx + (ln&3);
    v8bf a0, a1;
    __builtin_memcpy(&a0, Qs + srow*HD + 8*hi, 16);
    __builtin_memcpy(&a1, Qs + srow*HD + 16 + 8*hi, 16);
    v16bf Aop = SHUF16(a0, a1);

    #pragma unroll
    for (int t = 0; t < 4; ++t) {                       // 4 N-tiles of the 8x8 key union
      const int u    = 16*t + ln;                        // union key index = B column
      const int krow = (bly + (u>>3))*KR + blx + (u&7);
      // B operand: column n = union key, K = channels {8*hi..8*hi+7, 16+8*hi..16+8*hi+7}
      v8bf b0, b1;
      __builtin_memcpy(&b0, Ks + krow*HD + 8*hi,      16);
      __builtin_memcpy(&b1, Ks + krow*HD + 16 + 8*hi, 16);
      v16bf Bop = SHUF16(b0, b1);
      v8f cc = {};
      cc = __builtin_amdgcn_wmma_f32_16x16x32_bf16(false, Aop, false, Bop, (short)0, cc, false, false);
      // D layout: lane ln holds col n, VGPR r holds row m = r + 8*hi
      #pragma unroll
      for (int r = 0; r < 8; ++r) Ssm[(r + 8*hi)*64 + u] = cc[r];
    }
    asm volatile("s_wait_dscnt 0" ::: "memory");        // intra-wave: score dump -> softmax reads

    if (lane < 16) {                                    // one source pixel per lane
      const int m = lane;
      const int lsy = bly + (m>>2), lsx = blx + (m&3);
      float sc[25]; float mx = -3.0e38f;
      #pragma unroll
      for (int dy = 0; dy < 5; ++dy)
        #pragma unroll
        for (int dx = 0; dx < 5; ++dx) {
          float v = Ssm[m*64 + ((m>>2)+dy)*8 + (m&3)+dx];  // u = (m/4+2+dy-2+? ) -> valid window map
          sc[dy*5+dx] = v; mx = fmaxf(mx, v);
        }
      float sum = 0.f;
      #pragma unroll
      for (int p = 0; p < 25; ++p) { sc[p] = __expf(sc[p] - mx); sum += sc[p]; }
      const int gy = ty0 - 2 + lsy, gx = tx0 - 2 + lsx;
      const float mk = (gy>=0 && gy<HH && gx>=0 && gx<WW) ? Mp[(size_t)nb*HWSZ + gy*WW + gx] : 0.f;
      const float rs = mk / sum;
      __bf16* arow = At + (lsy*SR + lsx)*ATT_STRIDE;
      #pragma unroll
      for (int p = 0; p < 25; ++p) arow[p] = (__bf16)(sc[p] * rs);
    }
  }
  __syncthreads();

  // ---------------- Phase 2: diffusion out = W * V per 4x4 output block (WMMA) ----------------
  float* Outs = (float*)(smem + OFF_S);                 // 32ch x 16x16 f32 staging (aliases Ssm)
  for (int ob = wv; ob < 16; ob += 8) {
    const int oby = (ob >> 2) * 4, obx = (ob & 3) * 4;
    const int omy = ln >> 2, omx = ln & 3;
    v8f acc0 = {}, acc1 = {};
    #pragma unroll
    for (int t = 0; t < 2; ++t) {                       // K dim = 64 union sources, 2 chunks of 32
      // A operand: W[o][s] = att[s][o-s] (0 outside 5x5 window), gathered per A-layout
      v16bf Aw;
      #pragma unroll
      for (int e = 0; e < 16; ++e) {
        const int kl = (e < 8) ? (8*hi + e) : (16 + 8*hi + (e - 8));
        const int up = 32*t + kl;                       // union source index
        const int uy = up >> 3, ux = up & 7;
        const int py = omy - uy + 4, px = omx - ux + 4; // p = (dy+2, dx+2)
        __bf16 w = (__bf16)0.f;
        if (py >= 0 && py <= 4 && px >= 0 && px <= 4)
          w = At[((oby+uy)*SR + obx+ux)*ATT_STRIDE + py*5 + px];
        Aw[e] = w;
      }
      // B operand: V[s][c]; per lane K rows = sources (two contiguous x-runs), col = channel
      const int ls0 = (oby + 4*t + hi)*SR + obx;
      {
        v8bf b0, b1;
        __builtin_memcpy(&b0, Vt + ln*NSRC + ls0,      16);
        __builtin_memcpy(&b1, Vt + ln*NSRC + ls0 + 2*SR, 16);
        v16bf Bop = SHUF16(b0, b1);
        acc0 = __builtin_amdgcn_wmma_f32_16x16x32_bf16(false, Aw, false, Bop, (short)0, acc0, false, false);
      }
      {
        v8bf b0, b1;
        __builtin_memcpy(&b0, Vt + (16+ln)*NSRC + ls0,      16);
        __builtin_memcpy(&b1, Vt + (16+ln)*NSRC + ls0 + 2*SR, 16);
        v16bf Bop = SHUF16(b0, b1);
        acc1 = __builtin_amdgcn_wmma_f32_16x16x32_bf16(false, Aw, false, Bop, (short)0, acc1, false, false);
      }
    }
    #pragma unroll
    for (int r = 0; r < 8; ++r) {
      const int m = r + 8*hi;
      const int y = oby + (m>>2), x = obx + (m&3);
      Outs[ln*(TS*TS)      + y*TS + x] = acc0[r];       // channel ln
      Outs[(16+ln)*(TS*TS) + y*TS + x] = acc1[r];       // channel 16+ln
    }
  }
  __syncthreads();

  // ---------------- Phase 3: coalesced f32 writeback ----------------
  float* Og = Op + (size_t)(nb*DD + head*HD) * HWSZ;
  for (int e = tid; e < HD*TS*TS; e += 256) {
    const int x = e & 15, y = (e >> 4) & 15, c = e >> 8;
    Og[(size_t)c*HWSZ + (ty0+y)*WW + tx0+x] = Outs[c*(TS*TS) + y*TS + x];
  }
}

extern "C" void kernel_launch(void* const* d_in, const int* in_sizes, int n_in,
                              void* d_out, int out_size, void* d_ws, size_t ws_size,
                              hipStream_t stream) {
  // setup_inputs order: V, K, Q, ksize, dilation, mask. ksize=5 / dilation=1 are baked in.
  const float* Vp = (const float*)d_in[0];
  const float* Kp = (const float*)d_in[1];
  const float* Qp = (const float*)d_in[2];
  const float* Mp = (const float*)d_in[5];
  float* Op = (float*)d_out;
  (void)in_sizes; (void)n_in; (void)out_size; (void)d_ws; (void)ws_size;

  dim3 grid(NB * NHEADS * (HH/TS) * (WW/TS));   // 4096 workgroups
  natt_diffuse_kernel<<<grid, 256, SMEM_BYTES, stream>>>(Vp, Kp, Qp, Mp, Op);
}
